// QattenMixer_63333587746988
// MI455X (gfx1250) — matrix operand order repaired and checked by reading, not compiled, not run on hardware
//
#include <hip/hip_runtime.h>
#include <hip/hip_bf16.h>

typedef __attribute__((ext_vector_type(16))) _Float16 v16h;
typedef __attribute__((ext_vector_type(8)))  float    v8f;

// Problem constants
constexpr int kBS = 128, kT = 32;
constexpr int kB  = kBS * kT;      // 4096 batch rows
constexpr int kN  = 16;            // agents
constexpr int kU  = 32;            // unit feature dim
constexpr int kS  = 560;           // state dim
constexpr int kH  = 4;             // heads
constexpr int kE  = 32;            // embed dim
constexpr int kGH = 64;            // graph hidden
constexpr float kCOEF = 0.001f;
constexpr float kNEG  = -99999999.0f;
constexpr float kInvSqrtE = 0.17677669529663687f; // 1/sqrt(32)

constexpr int GSTR = 68;   // padded row stride (floats) for s_A/s_Bm: 272B, 16B-aligned,
                           // banks fan out by 4 per row -> conflict-free b128 loads
constexpr int PSTR = 20;   // padded row stride for s_part: 80B, 16B-aligned, halves disjoint

// Output layout (flat, in reference return order)
constexpr int Q_OFF   = 0;            // q_tot: 4096
constexpr int REG_OFF = kB;           // attend_mag_regs: 1
constexpr int ENT_OFF = kB + 1;       // head_entropies: 4
constexpr int ADJ_OFF = kB + 1 + kH;  // adj: 4096*256

__global__ void qatten_zero_ws(float* ws) {
    if (threadIdx.x < 5) ws[threadIdx.x] = 0.f;
}

__device__ __forceinline__ float dot4(const float4 a, const float4 b) {
    return a.x * b.x + a.y * b.y + a.z * b.z + a.w * b.w;
}

__global__ __launch_bounds__(128)
void qatten_main(const float* __restrict__ agent_qs,
                 const float* __restrict__ states,
                 const int*   __restrict__ actions,
                 const float* __restrict__ sel_W,
                 const float* __restrict__ key_W,
                 const float* __restrict__ wh_W,
                 const float* __restrict__ wh_b,
                 const float* __restrict__ V_W1,
                 const float* __restrict__ V_b1,
                 const float* __restrict__ V_W2,
                 const float* __restrict__ V_b2,
                 const float* __restrict__ g_W1,
                 const float* __restrict__ g_b1,
                 const float* __restrict__ g_W2,
                 const float* __restrict__ g_b2,
                 const float* __restrict__ g_W3,
                 const float* __restrict__ g_b3,
                 float* __restrict__ out,
                 float* __restrict__ ws)
{
    __shared__ __attribute__((aligned(16))) float s_st[kS];        // state row
    __shared__ float s_sel[kH * kE];
    __shared__ float s_m[kH * kU];
    __shared__ float s_logits[kH * kN];
    __shared__ float s_attw[kH * kN];
    __shared__ __attribute__((aligned(16))) float s_A[kN * GSTR];  // unit@W1a^T + b1
    __shared__ __attribute__((aligned(16))) float s_Bm[kN * GSTR]; // unit@W1b^T
    __shared__ __attribute__((aligned(16))) float s_part[256 * PSTR]; // layer-3 partials
    __shared__ float s_col[kN];
    __shared__ float s_wh[kH];
    __shared__ float s_v;
    __shared__ float s_red[128];

    const int b   = blockIdx.x;
    const int tid = threadIdx.x;

    // ---- load state row (float4) ----
    {
        const float4* src = reinterpret_cast<const float4*>(states + (size_t)b * kS);
        float4* dst = reinterpret_cast<float4*>(s_st);
        for (int i = tid; i < kS / 4; i += 128) dst[i] = src[i];
    }
    __syncthreads();

    const float4* st4 = reinterpret_cast<const float4*>(s_st);

    // ---- sel[h,e] = st . sel_W[h,e,:]  (one (h,e) per thread, float4) ----
    {
        const float4* w4 = reinterpret_cast<const float4*>(sel_W + (size_t)tid * kS);
        float acc = 0.f;
        for (int s = 0; s < kS / 4; ++s) acc += dot4(st4[s], w4[s]);
        s_sel[tid] = acc;
    }
    __syncthreads();

    // ---- m[h,u] = sum_e sel[h,e] * key_W[h,e,u] ----
    {
        const int h = tid >> 5, u = tid & 31;
        float acc = 0.f;
        for (int e = 0; e < kE; ++e)
            acc += s_sel[h * kE + e] * key_W[(h * kE + e) * kU + u];
        s_m[tid] = acc;
    }
    __syncthreads();

    // ---- logits / mask (tid<64), w_head (64..67), V hidden (96..127) ----
    if (tid < 64) {
        const int h = tid >> 4, n = tid & 15;
        float acc = 0.f;
        for (int u = 0; u < kU; ++u) acc += s_st[n * kU + u] * s_m[h * kU + u];
        s_logits[tid] = acc;
        float sc = acc * kInvSqrtE;
        if (actions[b * kN + n] == 0) sc = kNEG;
        s_attw[tid] = sc;
    } else if (tid < 68) {
        const int h = tid - 64;
        const float4* w4 = reinterpret_cast<const float4*>(wh_W + h * kS);
        float acc = wh_b[h];
        for (int s = 0; s < kS / 4; ++s) acc += dot4(st4[s], w4[s]);
        s_wh[h] = fabsf(acc);
    } else if (tid >= 96) {
        const int e = tid - 96;
        const float4* w4 = reinterpret_cast<const float4*>(V_W1 + e * kS);
        float acc = V_b1[e];
        for (int s = 0; s < kS / 4; ++s) acc += dot4(st4[s], w4[s]);
        s_red[e] = fmaxf(acc, 0.f) * V_W2[e];
    }
    __syncthreads();

    // ---- softmax per head / reg partial / V finalize ----
    if (tid < 4) {
        const int h = tid;
        float mx = -1e30f;
        for (int n = 0; n < 16; ++n) mx = fmaxf(mx, s_attw[h * 16 + n]);
        float ex[16], sum = 0.f;
        for (int n = 0; n < 16; ++n) { ex[n] = __expf(s_attw[h * 16 + n] - mx); sum += ex[n]; }
        const float inv = 1.f / sum;
        for (int n = 0; n < 16; ++n) s_attw[h * 16 + n] = ex[n] * inv;
    } else if (tid < 8) {
        const int h = tid - 4;
        float acc = 0.f;
        for (int n = 0; n < 16; ++n) { float l = s_logits[h * 16 + n]; acc += l * l; }
        s_red[32 + h] = acc;
    } else if (tid == 8) {
        float acc = V_b2[0];
        for (int e = 0; e < 32; ++e) acc += s_red[e];
        s_v = acc;
    }
    __syncthreads();
    if (tid == 0)
        atomicAdd(&ws[0], s_red[32] + s_red[33] + s_red[34] + s_red[35]);

    // ---- GraphLearner layer-1 rank structure: A'[i,g] (+b1), Bm[j,g] ----
    for (int idx = tid; idx < 2 * kN * kGH; idx += 128) {
        const int isB = idx >> 10;
        const int i = (idx & 1023) >> 6;
        const int g = idx & 63;
        const float4* w4 = reinterpret_cast<const float4*>(g_W1 + g * (2 * kU) + (isB ? kU : 0));
        const float4* u4 = reinterpret_cast<const float4*>(&s_st[i * kU]);
        float acc = isB ? 0.f : g_b1[g];
        for (int u = 0; u < kU / 4; ++u) acc += dot4(u4[u], w4[u]);
        if (isB) s_Bm[i * GSTR + g] = acc; else s_A[i * GSTR + g] = acc;
    }
    if (tid < 16) s_col[tid] = 0.f;
    __syncthreads();

    // ---- layer 2 GEMM on WMMA, layer-3 partials to LDS (no shuffles) ----
    const int wave   = tid >> 5;
    const int lane   = tid & 31;
    const int lane16 = lane & 15;
    const bool hi    = lane >= 16;
    const int m_row  = hi ? lane - 16 : lane;   // A-fragment row within tile

    // B fragments: W2^T (K=g, N=o), per ISA 16-bit B layout; 4x b128 per fragment
    v16h bfr[4][2];
    float w3v[4], b2v[4];
    const float4* W24 = reinterpret_cast<const float4*>(g_W2);
#pragma unroll
    for (int nt = 0; nt < 4; ++nt) {
        const int o = nt * 16 + lane16;
        w3v[nt] = g_W3[o];
        b2v[nt] = g_b2[o];
#pragma unroll
        for (int kc = 0; kc < 2; ++kc) {
            const int base = (o * kGH + kc * 32 + (hi ? 8 : 0)) >> 2;
            const float4 q0 = W24[base + 0], q1 = W24[base + 1];
            const float4 q2 = W24[base + 4], q3 = W24[base + 5];
            v16h bf;
            bf[0]=(_Float16)q0.x;  bf[1]=(_Float16)q0.y;  bf[2]=(_Float16)q0.z;  bf[3]=(_Float16)q0.w;
            bf[4]=(_Float16)q1.x;  bf[5]=(_Float16)q1.y;  bf[6]=(_Float16)q1.z;  bf[7]=(_Float16)q1.w;
            bf[8]=(_Float16)q2.x;  bf[9]=(_Float16)q2.y;  bf[10]=(_Float16)q2.z; bf[11]=(_Float16)q2.w;
            bf[12]=(_Float16)q3.x; bf[13]=(_Float16)q3.y; bf[14]=(_Float16)q3.z; bf[15]=(_Float16)q3.w;
            bfr[nt][kc] = bf;
        }
    }

    for (int mt = wave; mt < 16; mt += 4) {   // M-tile = pairs (i=mt, j=row)
        v8f acc[4];
#pragma unroll
        for (int nt = 0; nt < 4; ++nt)
            acc[nt] = (v8f){0.f, 0.f, 0.f, 0.f, 0.f, 0.f, 0.f, 0.f};

        const float4* A4 = reinterpret_cast<const float4*>(&s_A[mt * GSTR]);
        const float4* B4 = reinterpret_cast<const float4*>(&s_Bm[m_row * GSTR]);

#pragma unroll
        for (int kc = 0; kc < 2; ++kc) {
            // ISA A-layout K indices: two runs of 8 consecutive -> 4x b128 each side
            const int i0 = (kc * 32 + (hi ? 8 : 0)) >> 2;
            const float4 a0 = A4[i0], a1 = A4[i0 + 1], a2 = A4[i0 + 4], a3 = A4[i0 + 5];
            const float4 c0 = B4[i0], c1 = B4[i0 + 1], c2 = B4[i0 + 4], c3 = B4[i0 + 5];
            v16h af;
            af[0]=(_Float16)fmaxf(a0.x+c0.x,0.f);  af[1]=(_Float16)fmaxf(a0.y+c0.y,0.f);
            af[2]=(_Float16)fmaxf(a0.z+c0.z,0.f);  af[3]=(_Float16)fmaxf(a0.w+c0.w,0.f);
            af[4]=(_Float16)fmaxf(a1.x+c1.x,0.f);  af[5]=(_Float16)fmaxf(a1.y+c1.y,0.f);
            af[6]=(_Float16)fmaxf(a1.z+c1.z,0.f);  af[7]=(_Float16)fmaxf(a1.w+c1.w,0.f);
            af[8]=(_Float16)fmaxf(a2.x+c2.x,0.f);  af[9]=(_Float16)fmaxf(a2.y+c2.y,0.f);
            af[10]=(_Float16)fmaxf(a2.z+c2.z,0.f); af[11]=(_Float16)fmaxf(a2.w+c2.w,0.f);
            af[12]=(_Float16)fmaxf(a3.x+c3.x,0.f); af[13]=(_Float16)fmaxf(a3.y+c3.y,0.f);
            af[14]=(_Float16)fmaxf(a3.z+c3.z,0.f); af[15]=(_Float16)fmaxf(a3.w+c3.w,0.f);
#pragma unroll
            for (int nt = 0; nt < 4; ++nt)
                acc[nt] = __builtin_amdgcn_wmma_f32_16x16x32_f16(
                    false, af, false, bfr[nt][kc], (short)0, acc[nt], false, false);
        }

        // layer-3 column partials: each lane stores its 4-column contribution
#pragma unroll
        for (int v = 0; v < 8; ++v) {
            const int m = v + (hi ? 8 : 0);
            float part = 0.f;
#pragma unroll
            for (int nt = 0; nt < 4; ++nt) {
                const float h2 = fmaxf(acc[nt][v] + b2v[nt], 0.f);
                part += h2 * w3v[nt];
            }
            s_part[(mt * 16 + m) * PSTR + lane16] = part;  // fire-and-forget
        }
    }
    __syncthreads();

    // ---- reduce partials -> el -> adjacency + column sums ----
    const float b3 = g_b3[0];
    for (int p = tid; p < 256; p += 128) {
        const int i = p >> 4, j = p & 15;
        const float4* r = reinterpret_cast<const float4*>(&s_part[p * PSTR]);
        const float4 x0 = r[0], x1 = r[1], x2 = r[2], x3 = r[3];
        const float el = (x0.x+x0.y+x0.z+x0.w) + (x1.x+x1.y+x1.z+x1.w)
                       + (x2.x+x2.y+x2.z+x2.w) + (x3.x+x3.y+x3.z+x3.w) + b3;
        const float a = 1.f / (1.f + __expf(-el)) + (i == j ? 1.f : 0.f);
        out[ADJ_OFF + (size_t)b * 256 + p] = a;
        atomicAdd(&s_col[j], a);
    }
    __syncthreads();

    // ---- adj-weighted renorm, head_q, entropy, q_tot ----
    if (tid < 64) {
        const int n = tid & 15;
        s_red[tid] = s_attw[tid] * (s_col[n] * 0.0625f);
    }
    __syncthreads();
    if (tid < 4) {
        const int h = tid;
        float sum = 0.f;
        for (int n = 0; n < 16; ++n) sum += s_red[h * 16 + n];
        const float inv = 1.f / (sum + 1e-8f);
        float hq = 0.f, ent = 0.f;
        for (int n = 0; n < 16; ++n) {
            const float w = s_red[h * 16 + n] * inv;
            hq  += agent_qs[b * kN + n] * w;
            ent -= logf(w + 1e-8f) * w;
        }
        s_red[64 + h] = s_wh[h] * hq;
        atomicAdd(&ws[1 + h], ent);
    }
    __syncthreads();
    if (tid == 0)
        out[Q_OFF + b] = s_red[64] + s_red[65] + s_red[66] + s_red[67] + s_v;
}

__global__ void qatten_finalize(const float* __restrict__ ws, float* __restrict__ out) {
    const int t = threadIdx.x;
    if (t == 0)      out[REG_OFF]         = kCOEF * ws[0] / (float)(kB * kN);
    else if (t < 5)  out[ENT_OFF + t - 1] = ws[t] / (float)kB;
}

extern "C" void kernel_launch(void* const* d_in, const int* in_sizes, int n_in,
                              void* d_out, int out_size, void* d_ws, size_t ws_size,
                              hipStream_t stream) {
    const float* agent_qs = (const float*)d_in[0];
    const float* states   = (const float*)d_in[1];
    const int*   actions  = (const int*)  d_in[2];
    const float* sel_W    = (const float*)d_in[3];
    const float* key_W    = (const float*)d_in[4];
    const float* wh_W     = (const float*)d_in[5];
    const float* wh_b     = (const float*)d_in[6];
    const float* V_W1     = (const float*)d_in[7];
    const float* V_b1     = (const float*)d_in[8];
    const float* V_W2     = (const float*)d_in[9];
    const float* V_b2     = (const float*)d_in[10];
    const float* g_W1     = (const float*)d_in[11];
    const float* g_b1     = (const float*)d_in[12];
    const float* g_W2     = (const float*)d_in[13];
    const float* g_b2     = (const float*)d_in[14];
    const float* g_W3     = (const float*)d_in[15];
    const float* g_b3     = (const float*)d_in[16];

    float* out = (float*)d_out;
    float* ws  = (float*)d_ws;

    qatten_zero_ws<<<1, 32, 0, stream>>>(ws);
    qatten_main<<<kB, 128, 0, stream>>>(agent_qs, states, actions, sel_W, key_W,
                                        wh_W, wh_b, V_W1, V_b1, V_W2, V_b2,
                                        g_W1, g_b1, g_W2, g_b2, g_W3, g_b3,
                                        out, ws);
    qatten_finalize<<<1, 32, 0, stream>>>(ws, out);
}
